// Decoder_34265249087980
// MI455X (gfx1250) — compile-verified, matrix operand order checked by reading, not run on hardware
//
#include <hip/hip_runtime.h>

// ---------------------------------------------------------------------------
// MI455X (gfx1250) persistent attention-LSTM decoder.
//   * All GEMMs use v_wmma_f32_16x16x32_bf16 (M=16 == batch == one tile row).
//   * Weights pre-swizzled once into WMMA B-fragment layout (bf16), resident
//     in the 192MB L2 across the whole 96-step recurrence.
//   * Activation staging uses GLOBAL_LOAD_ASYNC_TO_LDS_B128 (+ s_wait_asynccnt)
//     so A-operands flow memory->LDS without VGPR round-trips.
//   * One persistent kernel (16 WGs x 256 thr = 128 wave32 = 128 N-tiles of
//     the 2048-wide LSTM gate GEMMs), grid-synced with a monotonic counter.
// ---------------------------------------------------------------------------

typedef __bf16  bf16;
typedef bf16    v16bf __attribute__((ext_vector_type(16)));
typedef float   v8f   __attribute__((ext_vector_type(8)));
typedef unsigned v8u  __attribute__((ext_vector_type(8)));

#define AS3 __attribute__((address_space(3)))

#define B_    16
#define T_    96
#define F_    512
#define H_    512
#define NM_   80
#define KT0   35     // 1120/32 K-tiles, LSTM0
#define KT1   32     // 1024/32 K-tiles, LSTM1
#define NWG   16

// ------------------------- bf16 helpers ------------------------------------
__device__ __forceinline__ unsigned bfbits(float f) {
  unsigned u = __float_as_uint(f);
  unsigned r = u + 0x7fffu + ((u >> 16) & 1u);   // round-to-nearest-even
  return r >> 16;
}
__device__ __forceinline__ unsigned pack_bf2(float lo, float hi) {
#if defined(__has_builtin) && __has_builtin(__builtin_amdgcn_cvt_pk_bf16_f32)
  typedef __bf16 v2bf __attribute__((ext_vector_type(2)));
  v2bf p = __builtin_amdgcn_cvt_pk_bf16_f32(lo, hi);
  return __builtin_bit_cast(unsigned, p);
#else
  return bfbits(lo) | (bfbits(hi) << 16);
#endif
}
__device__ __forceinline__ float sigm(float x) { return 1.f / (1.f + __expf(-x)); }

__device__ __forceinline__ v8f wmma_bf16(v16bf a, v16bf b, v8f c) {
  return __builtin_amdgcn_wmma_f32_16x16x32_bf16(false, a, false, b, (short)0, c,
                                                 false, false);
}

// ------------------------- async global -> LDS (CDNA5 TDM-lite path) -------
__device__ __forceinline__ void async_ld_b128(unsigned lds_off,
                                              const unsigned* gsrc) {
  // GLOBAL_LOAD_ASYNC_TO_LDS_B128: vdst = LDS byte offset, vaddr = 64b global.
  asm volatile("global_load_async_to_lds_b128 %0, %1, off"
               :: "v"(lds_off), "v"(gsrc) : "memory");
}
__device__ __forceinline__ void wait_async0() {
#if defined(__has_builtin) && __has_builtin(__builtin_amdgcn_s_wait_asynccnt)
  __builtin_amdgcn_s_wait_asynccnt(0);
#else
  asm volatile("s_wait_asynccnt 0" ::: "memory");
#endif
}

// A fragment (16x32 bf16) from a packed-bf16 row-major buffer (dword units).
// lane<16: row=lane,    K = k0+0..7  and k0+16..23
// lane>=16: row=lane-16, K = k0+8..15 and k0+24..31   (per CDNA5 ISA 7.12.2)
__device__ __forceinline__ v16bf load_A_dw(const unsigned* base, int pitch_dw,
                                           int k0, int lane) {
  int row = lane & 15, kh = lane >> 4;
  const unsigned* p = base + row * pitch_dw + ((k0 + kh * 8) >> 1);
  v8u d;
#pragma unroll
  for (int j = 0; j < 4; ++j) { d[j] = p[j]; d[4 + j] = p[8 + j]; }
  return __builtin_bit_cast(v16bf, d);
}

// A fragment from fp32 row-major (convert on the fly).
__device__ __forceinline__ v16bf load_A_f32(const float* Abase, int pitch,
                                            int k0, int lane) {
  int row = lane & 15, kh = lane >> 4;
  const float* p = Abase + row * pitch + k0 + kh * 8;
  v8u d;
#pragma unroll
  for (int j = 0; j < 4; ++j) {
    d[j]     = pack_bf2(p[2 * j],      p[2 * j + 1]);
    d[4 + j] = pack_bf2(p[16 + 2 * j], p[16 + 2 * j + 1]);
  }
  return __builtin_bit_cast(v16bf, d);
}

// A fragment = 0.5*(h0+h1)  (query = mean over 2 LSTM layers)
__device__ __forceinline__ v16bf load_A_qavg(const float* h0, const float* h1,
                                             int k0, int lane) {
  int row = lane & 15, kh = lane >> 4;
  const float* p0 = h0 + row * H_ + k0 + kh * 8;
  const float* p1 = h1 + row * H_ + k0 + kh * 8;
  v8u d;
#pragma unroll
  for (int j = 0; j < 4; ++j) {
    d[j]     = pack_bf2(0.5f * (p0[2*j]    + p1[2*j]),
                        0.5f * (p0[2*j+1]  + p1[2*j+1]));
    d[4 + j] = pack_bf2(0.5f * (p0[16+2*j]   + p1[16+2*j]),
                        0.5f * (p0[16+2*j+1] + p1[16+2*j+1]));
  }
  return __builtin_bit_cast(v16bf, d);
}

// B fragment from pre-swizzled weights: 8 contiguous dwords per lane.
__device__ __forceinline__ v16bf load_B(const unsigned* wb, int tile, int lane) {
  const unsigned* p = wb + (((size_t)tile * 32 + lane) << 3);
  v8u d;
#pragma unroll
  for (int j = 0; j < 8; ++j) d[j] = p[j];
  return __builtin_bit_cast(v16bf, d);
}

// ------------------------- grid barrier ------------------------------------
__device__ __forceinline__ void grid_barrier(unsigned* cnt, unsigned& rnd) {
  __syncthreads();
  if (threadIdx.x == 0) {
    __threadfence();
    atomicAdd(cnt, 1u);
    unsigned target = (rnd + 1u) * gridDim.x;
    while (__atomic_load_n(cnt, __ATOMIC_RELAXED) < target)
      __builtin_amdgcn_s_sleep(1);
  }
  __syncthreads();
  __threadfence();
  rnd += 1u;
}

// ------------------------- init --------------------------------------------
__global__ void init_zero(unsigned* p, int n) {
  int i = blockIdx.x * 256 + threadIdx.x;
  if (i < n) p[i] = 0u;
}

// ------------------------- weight swizzle ----------------------------------
__device__ __forceinline__ float w0_at(const float* Wih0, const float* Whh0,
                                       int n, int k) {
  if (k < 592) return Wih0[n * 592 + k];
  if (k < 608) return 0.f;
  return Whh0[n * 512 + (k - 608)];
}
__device__ __forceinline__ float w1_at(const float* Wih1, const float* Whh1,
                                       int n, int k) {
  return (k < 512) ? Wih1[n * 512 + k] : Whh1[n * 512 + (k - 512)];
}

// tiles: wB0 128x35 | wB1 128x32 | wBa 32x16 | wBu 32x16 | wBp 10x16
__global__ void prep_swizzle(const float* __restrict__ Wih0, const float* __restrict__ Whh0,
                             const float* __restrict__ Wih1, const float* __restrict__ Whh1,
                             const float* __restrict__ Wa,   const float* __restrict__ Ua,
                             const float* __restrict__ Pj,
                             unsigned* __restrict__ wB0, unsigned* __restrict__ wB1,
                             unsigned* __restrict__ wBa, unsigned* __restrict__ wBu,
                             unsigned* __restrict__ wBp) {
  int gid = blockIdx.x * 256 + threadIdx.x;
  int tile = gid >> 5, lane = gid & 31;
  if (tile >= 9760) return;
  unsigned* outp; int nt, kt, mat;
  if (tile < 4480)      { mat = 0; nt = tile / 35; kt = tile % 35; outp = wB0 + (size_t)tile * 256; }
  else if (tile < 8576) { int li = tile - 4480; mat = 1; nt = li / 32; kt = li % 32; outp = wB1 + (size_t)li * 256; }
  else if (tile < 9088) { int li = tile - 8576; mat = 2; nt = li / 16; kt = li % 16; outp = wBa + (size_t)li * 256; }
  else if (tile < 9600) { int li = tile - 9088; mat = 3; nt = li / 16; kt = li % 16; outp = wBu + (size_t)li * 256; }
  else                  { int li = tile - 9600; mat = 4; nt = li / 16; kt = li % 16; outp = wBp + (size_t)li * 256; }
  int n  = nt * 16 + (lane & 15);
  int kh = lane >> 4;
#pragma unroll
  for (int v = 0; v < 8; ++v) {
    int k = kt * 32 + kh * 8 + ((v < 4) ? (2 * v) : (16 + 2 * (v - 4)));
    float lo, hi;
    if (mat == 0)      { lo = w0_at(Wih0, Whh0, n, k); hi = w0_at(Wih0, Whh0, n, k + 1); }
    else if (mat == 1) { lo = w1_at(Wih1, Whh1, n, k); hi = w1_at(Wih1, Whh1, n, k + 1); }
    else if (mat == 2) { lo = Wa[n * 512 + k]; hi = Wa[n * 512 + k + 1]; }
    else if (mat == 3) { lo = Ua[n * 512 + k]; hi = Ua[n * 512 + k + 1]; }
    else               { lo = Pj[n * 512 + k]; hi = Pj[n * 512 + k + 1]; }
    outp[lane * 8 + v] = pack_bf2(lo, hi);
  }
}

// ------------------------- UaK = enc @ Ua^T + Ua_b -------------------------
__global__ void __launch_bounds__(256) uak_kernel(const float* __restrict__ enc,
                                                  const unsigned* __restrict__ wBu,
                                                  const float* __restrict__ Ua_b,
                                                  float* __restrict__ UaK) {
  int wave = (blockIdx.x * 256 + threadIdx.x) >> 5;   // 0..3071
  int lane = threadIdx.x & 31;
  int mt = wave >> 5;     // 96 M-tiles over B*T=1536 rows
  int nt = wave & 31;     // 32 N-tiles over H
  v8f acc = {0.f, 0.f, 0.f, 0.f, 0.f, 0.f, 0.f, 0.f};
  const float* Abase = enc + (size_t)mt * 16 * F_;
  for (int kt = 0; kt < 16; ++kt) {
    v16bf a = load_A_f32(Abase, F_, kt * 32, lane);
    v16bf b = load_B(wBu, nt * 16 + kt, lane);
    acc = wmma_bf16(a, b, acc);
  }
  int mb = (lane >> 4) << 3;
  int n  = nt * 16 + (lane & 15);
  float bias = Ua_b[n];
#pragma unroll
  for (int v = 0; v < 8; ++v)
    UaK[(size_t)(mt * 16 + mb + v) * H_ + n] = acc[v] + bias;
}

// ------------------------- persistent decoder ------------------------------
__global__ void __launch_bounds__(256) decoder_kernel(
    const float* __restrict__ enc,
    const float* __restrict__ Wa_b, const float* __restrict__ Va_w,
    const float* __restrict__ Va_b,
    const float* __restrict__ bih0, const float* __restrict__ bhh0,
    const float* __restrict__ bih1, const float* __restrict__ bhh1,
    const float* __restrict__ projb,
    const unsigned* __restrict__ wB0, const unsigned* __restrict__ wB1,
    const unsigned* __restrict__ wBa, const unsigned* __restrict__ wBp,
    const float* __restrict__ UaK,
    float* __restrict__ h0g, float* __restrict__ h1g,
    float* __restrict__ c0g, float* __restrict__ c1g,
    float* __restrict__ gates, float* __restrict__ qW,
    float* __restrict__ scores,
    unsigned* __restrict__ xg,   // [16][560] dwords = packed bf16 [16][1120]
    unsigned* __restrict__ x1g,  // [16][512] dwords = packed bf16 [16][1024]
    unsigned* __restrict__ barcnt,
    float* __restrict__ out) {
  __shared__ __align__(16) unsigned ldsx[16 * 560];
  const int tid     = threadIdx.x;
  const int lane    = tid & 31;
  const int wv      = (blockIdx.x << 3) + (tid >> 5);  // 0..127 wave32 id
  const int gthread = blockIdx.x * 256 + tid;          // 0..4095
  const unsigned lbase = (unsigned)(size_t)(AS3 unsigned*)ldsx;  // LDS byte off
  unsigned bar_round = 0;

  for (int t = 0; t < T_; ++t) {
    // ---- A1: qW = mean(h0,h1) @ Wa^T + Wa_b  (32 N-tiles, 16 K-tiles) ----
    if (wv < 32) {
      v8f acc = {0.f, 0.f, 0.f, 0.f, 0.f, 0.f, 0.f, 0.f};
      for (int kt = 0; kt < 16; ++kt) {
        v16bf a = load_A_qavg(h0g, h1g, kt * 32, lane);
        v16bf b = load_B(wBa, wv * 16 + kt, lane);
        acc = wmma_bf16(a, b, acc);
      }
      int mb = (lane >> 4) << 3;
      int n  = wv * 16 + (lane & 15);
      float bias = Wa_b[n];
#pragma unroll
      for (int v = 0; v < 8; ++v) qW[(mb + v) * H_ + n] = acc[v] + bias;
    }
    grid_barrier(barcnt, bar_round);

    // ---- A2: scores[b,tt] = Va_b + sum_h Va[h]*tanh(qW+UaK) --------------
    {
      float vb = Va_b[0];
      for (int it = wv; it < B_ * T_; it += 128) {
        int b = it / T_, tt = it - b * T_;
        const float* qrow = qW + b * H_;
        const float* urow = UaK + (size_t)(b * T_ + tt) * H_;
        float acc = 0.f;
        for (int u = lane; u < H_; u += 32)
          acc += Va_w[u] * tanhf(qrow[u] + urow[u]);
#pragma unroll
        for (int off = 16; off > 0; off >>= 1) acc += __shfl_down(acc, off, 32);
        if (lane == 0) scores[b * T_ + tt] = acc + vb;
      }
    }
    grid_barrier(barcnt, bar_round);

    // ---- A3: softmax over T, ctx = w @ enc; stage ctx as bf16 pair -------
    {
      int b = gthread >> 8;
      int f = (gthread & 255) << 1;    // even f; thread owns (f, f+1)
      const float* sr = scores + b * T_;
      float mx = -1e30f;
      for (int u = 0; u < T_; ++u) mx = fmaxf(mx, sr[u]);
      float den = 0.f;
      for (int u = 0; u < T_; ++u) den += __expf(sr[u] - mx);
      float inv = 1.f / den;
      const float* eb = enc + (size_t)b * T_ * F_ + f;
      float ca = 0.f, cb = 0.f;
      for (int u = 0; u < T_; ++u) {
        float w = __expf(sr[u] - mx);
        ca += w * eb[u * F_];
        cb += w * eb[u * F_ + 1];
      }
      xg[b * 560 + ((NM_ + f) >> 1)] = pack_bf2(ca * inv, cb * inv);
    }
    grid_barrier(barcnt, bar_round);

    // ---- B: LSTM0 gates = [inp|ctx|pad|h0_prev] @ W0^T (128x35 tiles) ----
    // async DMA xg -> LDS (2240 x b128), no VGPR round-trip
    for (int i = tid; i < (16 * 560) / 4; i += 256)
      async_ld_b128(lbase + (unsigned)i * 16u, xg + i * 4);
    wait_async0();
    __syncthreads();
    {
      int nt = wv;
      v8f acc = {0.f, 0.f, 0.f, 0.f, 0.f, 0.f, 0.f, 0.f};
      for (int kt = 0; kt < KT0; ++kt) {
        if (kt + 1 < KT0)
          __builtin_prefetch(wB0 + (((size_t)(nt * KT0 + kt + 1) * 32 + lane) << 3), 0, 0);
        v16bf a = load_A_dw(ldsx, 560, kt * 32, lane);
        v16bf b = load_B(wB0, nt * KT0 + kt, lane);
        acc = wmma_bf16(a, b, acc);
      }
      int mb = (lane >> 4) << 3;
      int n  = nt * 16 + (lane & 15);
      float bias = bih0[n] + bhh0[n];
#pragma unroll
      for (int v = 0; v < 8; ++v) gates[(mb + v) * 2048 + n] = acc[v] + bias;
    }
    grid_barrier(barcnt, bar_round);

    // ---- B2: c0,h0 update; stage h0 (bf16) for W_hh0(next step) & LSTM1 --
    {
      int b = gthread >> 8;
      int u = (gthread & 255) << 1;
      const float* g = gates + b * 2048;
      float i0 = sigm(g[u]),            i1 = sigm(g[u + 1]);
      float f0 = sigm(g[512 + u]),      f1 = sigm(g[512 + u + 1]);
      float z0 = tanhf(g[1024 + u]),    z1 = tanhf(g[1024 + u + 1]);
      float o0 = sigm(g[1536 + u]),     o1 = sigm(g[1536 + u + 1]);
      float cc0 = f0 * c0g[b * H_ + u]     + i0 * z0;
      float cc1 = f1 * c0g[b * H_ + u + 1] + i1 * z1;
      float hh0 = o0 * tanhf(cc0), hh1 = o1 * tanhf(cc1);
      c0g[b * H_ + u] = cc0;  c0g[b * H_ + u + 1] = cc1;
      h0g[b * H_ + u] = hh0;  h0g[b * H_ + u + 1] = hh1;
      unsigned p = pack_bf2(hh0, hh1);
      xg [b * 560 + ((608 + u) >> 1)] = p;
      x1g[b * 512 + (u >> 1)]         = p;
    }
    grid_barrier(barcnt, bar_round);

    // ---- C: LSTM1 gates = [h0|h1_prev] @ W1^T (128x32 tiles) -------------
    for (int i = tid; i < (16 * 512) / 4; i += 256)
      async_ld_b128(lbase + (unsigned)i * 16u, x1g + i * 4);
    wait_async0();
    __syncthreads();
    {
      int nt = wv;
      v8f acc = {0.f, 0.f, 0.f, 0.f, 0.f, 0.f, 0.f, 0.f};
      for (int kt = 0; kt < KT1; ++kt) {
        if (kt + 1 < KT1)
          __builtin_prefetch(wB1 + (((size_t)(nt * KT1 + kt + 1) * 32 + lane) << 3), 0, 0);
        v16bf a = load_A_dw(ldsx, 512, kt * 32, lane);
        v16bf b = load_B(wB1, nt * KT1 + kt, lane);
        acc = wmma_bf16(a, b, acc);
      }
      int mb = (lane >> 4) << 3;
      int n  = nt * 16 + (lane & 15);
      float bias = bih1[n] + bhh1[n];
#pragma unroll
      for (int v = 0; v < 8; ++v) gates[(mb + v) * 2048 + n] = acc[v] + bias;
    }
    grid_barrier(barcnt, bar_round);

    // ---- C2: c1,h1 update; stage h1 (bf16) ------------------------------
    {
      int b = gthread >> 8;
      int u = (gthread & 255) << 1;
      const float* g = gates + b * 2048;
      float i0 = sigm(g[u]),            i1 = sigm(g[u + 1]);
      float f0 = sigm(g[512 + u]),      f1 = sigm(g[512 + u + 1]);
      float z0 = tanhf(g[1024 + u]),    z1 = tanhf(g[1024 + u + 1]);
      float o0 = sigm(g[1536 + u]),     o1 = sigm(g[1536 + u + 1]);
      float cc0 = f0 * c1g[b * H_ + u]     + i0 * z0;
      float cc1 = f1 * c1g[b * H_ + u + 1] + i1 * z1;
      float hh0 = o0 * tanhf(cc0), hh1 = o1 * tanhf(cc1);
      c1g[b * H_ + u] = cc0;  c1g[b * H_ + u + 1] = cc1;
      h1g[b * H_ + u] = hh0;  h1g[b * H_ + u + 1] = hh1;
      x1g[b * 512 + ((512 + u) >> 1)] = pack_bf2(hh0, hh1);
    }
    grid_barrier(barcnt, bar_round);

    // ---- D: out = h1 @ proj^T + proj_b (5 tiles); feed back as next inp --
    if (wv < 5) {
      v8f acc = {0.f, 0.f, 0.f, 0.f, 0.f, 0.f, 0.f, 0.f};
      for (int kt = 0; kt < 16; ++kt) {
        v16bf a = load_A_dw(x1g + 256, 512, kt * 32, lane);  // h1 half of x1g
        v16bf b = load_B(wBp, wv * 16 + kt, lane);
        acc = wmma_bf16(a, b, acc);
      }
      int mb = (lane >> 4) << 3;
      int n  = wv * 16 + (lane & 15);   // 0..79
      float bias = projb[n];
#pragma unroll
      for (int v = 0; v < 8; ++v) {
        float val = acc[v] + bias;
        int m = mb + v;                 // batch row
        out[(size_t)(m * T_ + t) * NM_ + n] = val;
        float nbv = __shfl_down(val, 1, 32);
        if ((lane & 1) == 0) xg[m * 560 + (n >> 1)] = pack_bf2(val, nbv);
      }
    }
    grid_barrier(barcnt, bar_round);
  }
}

// ------------------------- host launcher -----------------------------------
extern "C" void kernel_launch(void* const* d_in, const int* in_sizes, int n_in,
                              void* d_out, int out_size, void* d_ws, size_t ws_size,
                              hipStream_t stream) {
  const float* enc   = (const float*)d_in[0];
  const float* Wa_w  = (const float*)d_in[1];
  const float* Wa_b  = (const float*)d_in[2];
  const float* Ua_w  = (const float*)d_in[3];
  const float* Ua_b  = (const float*)d_in[4];
  const float* Va_w  = (const float*)d_in[5];
  const float* Va_b  = (const float*)d_in[6];
  const float* Wih0  = (const float*)d_in[7];
  const float* Whh0  = (const float*)d_in[8];
  const float* bih0  = (const float*)d_in[9];
  const float* bhh0  = (const float*)d_in[10];
  const float* Wih1  = (const float*)d_in[11];
  const float* Whh1  = (const float*)d_in[12];
  const float* bih1  = (const float*)d_in[13];
  const float* bhh1  = (const float*)d_in[14];
  const float* projw = (const float*)d_in[15];
  const float* projb = (const float*)d_in[16];
  float* out = (float*)d_out;

  char* ws = (char*)d_ws;
  size_t off = 0;
  auto alloc = [&](size_t bytes) -> void* {
    void* p = ws + off;
    off += (bytes + 255) & ~(size_t)255;
    return p;
  };
  // --- zero-initialized state region (must come first) ---
  unsigned* barcnt = (unsigned*)alloc(256);
  float* h0g   = (float*)alloc(16 * 512 * 4);
  float* h1g   = (float*)alloc(16 * 512 * 4);
  float* c0g   = (float*)alloc(16 * 512 * 4);
  float* c1g   = (float*)alloc(16 * 512 * 4);
  float* gates = (float*)alloc(16 * 2048 * 4);
  float* qW    = (float*)alloc(16 * 512 * 4);
  float* scores= (float*)alloc(16 * 96 * 4);
  unsigned* xg  = (unsigned*)alloc(16 * 560 * 4);
  unsigned* x1g = (unsigned*)alloc(16 * 512 * 4);
  int state_dwords = (int)(off / 4);
  // --- write-once regions ---
  unsigned* wB0 = (unsigned*)alloc((size_t)4480 * 1024);
  unsigned* wB1 = (unsigned*)alloc((size_t)4096 * 1024);
  unsigned* wBa = (unsigned*)alloc((size_t)512 * 1024);
  unsigned* wBu = (unsigned*)alloc((size_t)512 * 1024);
  unsigned* wBp = (unsigned*)alloc((size_t)160 * 1024);
  float* UaK    = (float*)alloc((size_t)1536 * 512 * 4);

  init_zero<<<(state_dwords + 255) / 256, 256, 0, stream>>>((unsigned*)ws,
                                                            state_dwords);
  prep_swizzle<<<1220, 256, 0, stream>>>(Wih0, Whh0, Wih1, Whh1, Wa_w, Ua_w,
                                         projw, wB0, wB1, wBa, wBu, wBp);
  uak_kernel<<<384, 256, 0, stream>>>(enc, wBu, Ua_b, UaK);
  decoder_kernel<<<NWG, 256, 0, stream>>>(
      enc, Wa_b, Va_w, Va_b, bih0, bhh0, bih1, bhh1, projb,
      wB0, wB1, wBa, wBp, UaK,
      h0g, h1g, c0g, c1g, gates, qW, scores, xg, x1g, barcnt, out);
}
